// QIDDM_PP_old_21285857919134
// MI455X (gfx1250) — matrix operand order repaired and checked by reading, not compiled, not run on hardware
//
#include <hip/hip_runtime.h>
#include <hip/hip_bf16.h>
#include <cstddef>

// ---------------------------------------------------------------------------
// QIDDM forward on MI455X (gfx1250, wave32).
//  - rank-16 subspace via block power iteration:
//      K-deep GEMMs use v_wmma_f32_16x16x32_bf16 (bf16 in, f32 accumulate)
//  - batchnorm + dense 16->8
//  - 4x 8-qubit statevector circuit (LDS, one block per sample)
//  - dense 8->16, back-projection (K=16) with v_wmma_f32_16x16x4_f32
// ---------------------------------------------------------------------------

typedef __attribute__((ext_vector_type(2)))  float  v2f;
typedef __attribute__((ext_vector_type(8)))  float  v8f;
typedef __attribute__((ext_vector_type(16))) __bf16 v16bf;

#define NB   2048          // batch
#define ND   4096          // flattened feature dim
#define NC   16            // subspace rank (2*HF)
#define NH   8             // HF
#define NPOW 8             // power-iteration sweeps

// ---------------- column mean over batch --------------------------------
__global__ __launch_bounds__(256) void k_colmean(const float* __restrict__ X,
                                                 float* __restrict__ mean) {
  int j = blockIdx.x * blockDim.x + threadIdx.x;
  if (j >= ND) return;
  float s = 0.f;
  for (int b = 0; b < NB; ++b) s += X[(size_t)b * ND + j];
  mean[j] = s * (1.f / (float)NB);
}

// ---------------- deterministic init of V (4096x16) ---------------------
__global__ __launch_bounds__(256) void k_initv(float* __restrict__ V) {
  int i = blockIdx.x * blockDim.x + threadIdx.x;
  if (i >= ND * NC) return;
  unsigned u = (unsigned)i * 2654435761u;
  u ^= u >> 16; u *= 2246822519u; u ^= u >> 13;
  V[i] = ((float)(u & 0xFFFFFFu) / 8388608.0f) - 1.0f;
}

// ---------------- Y[2048x16] = (X - mean) @ V[4096x16]  (bf16 WMMA) -----
// A 16x32 bf16 fragment: element e -> K = 8*half + e (e<8), 16+8*half+(e-8)
// B 32x16 bf16 fragment: element e -> K = 8*half + e (e<8), 16+8*half+(e-8), N = lane16
__global__ __launch_bounds__(256) void k_gemm_xc_v(const float* __restrict__ X,
                                                   const float* __restrict__ mean,
                                                   const float* __restrict__ V,
                                                   float* __restrict__ Y) {
  int wave = threadIdx.x >> 5;
  int lane = threadIdx.x & 31;
  int half = lane >> 4, l16 = lane & 15;
  int m0 = blockIdx.x * 128 + wave * 16;
  const float* xrow = X + (size_t)(m0 + l16) * ND;
  v8f acc = {};
  for (int k = 0; k < ND; k += 32) {
    __builtin_prefetch(xrow + k + 256, 0, 1);
    int ka = k + 8 * half;        // first run of 8 K values for A
    int kb = k + 8 * half;        // first run of 8 K values for B
    v16bf a, b;
#pragma unroll
    for (int e = 0; e < 8; ++e) {
      a[e]     = (__bf16)(xrow[ka + e]      - mean[ka + e]);
      a[8 + e] = (__bf16)(xrow[ka + 16 + e] - mean[ka + 16 + e]);
    }
#pragma unroll
    for (int e = 0; e < 16; ++e)
      b[e] = (__bf16)V[(kb + (e & 7) + ((e >> 3) << 4)) * NC + l16];
    acc = __builtin_amdgcn_wmma_f32_16x16x32_bf16(false, a, false, b,
                                                  (short)0, acc, false, false);
  }
#pragma unroll
  for (int v = 0; v < 8; ++v)
    Y[(m0 + v + 8 * half) * NC + l16] = acc[v];
}

// ---------------- W[4096x16] = (X - mean)^T @ Y[2048x16]  (bf16 WMMA) ---
__global__ __launch_bounds__(256) void k_gemm_xct_y(const float* __restrict__ X,
                                                    const float* __restrict__ mean,
                                                    const float* __restrict__ Y,
                                                    float* __restrict__ W) {
  int wave = threadIdx.x >> 5;
  int lane = threadIdx.x & 31;
  int half = lane >> 4, l16 = lane & 15;
  int m0 = blockIdx.x * 128 + wave * 16;          // feature index tile
  float mcol = mean[m0 + l16];
  v8f acc = {};
  for (int k = 0; k < NB; k += 32) {
    __builtin_prefetch(X + (size_t)(k + 32) * ND + m0 + l16, 0, 1);
    int ka = k + 8 * half;
    int kb = k + 8 * half;
    v16bf a, b;
#pragma unroll
    for (int e = 0; e < 8; ++e) {                 // A[m,k] = Xc[k, m0+l16]
      a[e]     = (__bf16)(X[(size_t)(ka + e)      * ND + m0 + l16] - mcol);
      a[8 + e] = (__bf16)(X[(size_t)(ka + 16 + e) * ND + m0 + l16] - mcol);
    }
#pragma unroll
    for (int e = 0; e < 16; ++e)
      b[e] = (__bf16)Y[(kb + (e & 7) + ((e >> 3) << 4)) * NC + l16];
    acc = __builtin_amdgcn_wmma_f32_16x16x32_bf16(false, a, false, b,
                                                  (short)0, acc, false, false);
  }
#pragma unroll
  for (int v = 0; v < 8; ++v)
    W[(m0 + v + 8 * half) * NC + l16] = acc[v];
}

// ---------------- modified Gram-Schmidt: V = orth(W), columns of 4096 ----
__global__ __launch_bounds__(256) void k_orthonormalize(const float* __restrict__ W,
                                                        float* __restrict__ V) {
  __shared__ float red[256];
  __shared__ float coef;
  int t = threadIdx.x;
  for (int c = 0; c < NC; ++c) {
    for (int d = t; d < ND; d += 256) V[d * NC + c] = W[d * NC + c];
    __syncthreads();
    for (int p = 0; p < c; ++p) {
      float part = 0.f;
      for (int d = t; d < ND; d += 256) part += V[d * NC + p] * V[d * NC + c];
      red[t] = part; __syncthreads();
      for (int off = 128; off > 0; off >>= 1) {
        if (t < off) red[t] += red[t + off];
        __syncthreads();
      }
      if (t == 0) coef = red[0];
      __syncthreads();
      float dot = coef;
      for (int d = t; d < ND; d += 256) V[d * NC + c] -= dot * V[d * NC + p];
      __syncthreads();
    }
    float part = 0.f;
    for (int d = t; d < ND; d += 256) { float v = V[d * NC + c]; part += v * v; }
    red[t] = part; __syncthreads();
    for (int off = 128; off > 0; off >>= 1) {
      if (t < off) red[t] += red[t + off];
      __syncthreads();
    }
    if (t == 0) coef = rsqrtf(fmaxf(red[0], 1e-20f));
    __syncthreads();
    float inv = coef;
    for (int d = t; d < ND; d += 256) V[d * NC + c] *= inv;
    __syncthreads();
  }
}

// ---------------- batchnorm stats over the 2048x16 projection -----------
__global__ __launch_bounds__(256) void k_bnstats(const float* __restrict__ Y,
                                                 float* __restrict__ stats) {
  __shared__ float ssum[256], ssq[256];
  int t = threadIdx.x;
  int c = t & 15, g = t >> 4;
  float s = 0.f, q = 0.f;
  for (int r = g; r < NB; r += 16) { float v = Y[r * NC + c]; s += v; q += v * v; }
  ssum[t] = s; ssq[t] = q; __syncthreads();
  for (int off = 128; off >= 16; off >>= 1) {
    if (t < off) { ssum[t] += ssum[t + off]; ssq[t] += ssq[t + off]; }
    __syncthreads();
  }
  if (t < 16) {
    float mu = ssum[t] * (1.f / (float)NB);
    float var = ssq[t] * (1.f / (float)NB) - mu * mu;
    stats[t] = mu;
    stats[16 + t] = var;
  }
}

// ---------------- BN + dense 16->8 (per-sample) --------------------------
__global__ __launch_bounds__(256) void k_bn_wd(const float* __restrict__ Y,
                                               const float* __restrict__ stats,
                                               const float* __restrict__ gamma,
                                               const float* __restrict__ beta,
                                               const float* __restrict__ Wd,
                                               const float* __restrict__ bd,
                                               float* __restrict__ xr8) {
  int b = blockIdx.x * blockDim.x + threadIdx.x;
  if (b >= NB) return;
  float xn[NC];
#pragma unroll
  for (int c = 0; c < NC; ++c) {
    float mu = stats[c], var = stats[16 + c];
    xn[c] = (Y[b * NC + c] - mu) * rsqrtf(var + 1e-5f) * gamma[c] + beta[c];
  }
#pragma unroll
  for (int j = 0; j < NH; ++j) {
    float acc = bd[j];
#pragma unroll
    for (int c = 0; c < NC; ++c) acc += xn[c] * Wd[j * NC + c];
    xr8[b * NH + j] = acc;
  }
}

// ---------------- one circuit pass: in/out (2048x8), one block/sample ----
__global__ __launch_bounds__(128) void k_circuit(const float* __restrict__ xin,
                                                 const float* __restrict__ w1,
                                                 float* __restrict__ xout,
                                                 int step) {
  __shared__ float sre[256], sim[256];
  __shared__ float gate[32][8];        // (i,l,j) -> {u00r,u00i,u01r,u01i,u10r,u10i,u11r,u11i}
  __shared__ float inp[8];
  int b = blockIdx.x;
  int t = threadIdx.x;

  if (t < 8) inp[t] = xin[b * NH + t];
  if (t < 32) {
    int i = t >> 4, l = (t >> 3) & 1, j = t & 7;
    const float* ww = w1 + (((step * 2 + i) * 2 + l) * NH + j) * 3;
    float phi = ww[0], th = ww[1], om = ww[2];
    float c = __cosf(0.5f * th), s = __sinf(0.5f * th);
    float ap = 0.5f * (phi + om), am = 0.5f * (phi - om);
    float ca = __cosf(ap), sa = __sinf(ap);
    float cm = __cosf(am), sm = __sinf(am);
    gate[t][0] =  ca * c;  gate[t][1] = -sa * c;   // u00 = e^{-i*ap} c
    gate[t][2] = -cm * s;  gate[t][3] = -sm * s;   // u01 = -e^{+i*am} s
    gate[t][4] =  cm * s;  gate[t][5] = -sm * s;   // u10 = e^{-i*am} s
    gate[t][6] =  ca * c;  gate[t][7] =  sa * c;   // u11 = e^{+i*ap} c
  }
  sre[t] = (t == 0) ? 1.f : 0.f;  sim[t] = 0.f;
  sre[t + 128] = 0.f;             sim[t + 128] = 0.f;
  __syncthreads();

  // RZ diagonal phases for the two amplitudes this thread owns
  float ph0 = 0.f, ph1 = 0.f;
  {
    int k0 = t, k1 = t + 128;
#pragma unroll
    for (int j = 0; j < NH; ++j) {
      float z0 = ((k0 >> (7 - j)) & 1) ? -1.f : 1.f;
      float z1 = ((k1 >> (7 - j)) & 1) ? -1.f : 1.f;
      ph0 += inp[j] * z0;
      ph1 += inp[j] * z1;
    }
    ph0 *= -0.5f; ph1 *= -0.5f;
  }
  float c0 = __cosf(ph0), s0 = __sinf(ph0);
  float c1 = __cosf(ph1), s1 = __sinf(ph1);

  for (int i = 0; i < 2; ++i) {
    // diagonal RZ
    {
      float r = sre[t], q = sim[t];
      sre[t] = r * c0 - q * s0;  sim[t] = r * s0 + q * c0;
      r = sre[t + 128]; q = sim[t + 128];
      sre[t + 128] = r * c1 - q * s1;  sim[t + 128] = r * s1 + q * c1;
    }
    __syncthreads();
    for (int l = 0; l < 2; ++l) {
      for (int j = 0; j < NH; ++j) {
        int s = 7 - j;                                      // bit position of wire j
        int i0 = ((t >> s) << (s + 1)) | (t & ((1 << s) - 1));
        int i1 = i0 | (1 << s);
        const float* g = gate[(i * 2 + l) * NH + j];
        float a0r = sre[i0], a0i = sim[i0];
        float a1r = sre[i1], a1i = sim[i1];
        float n0r = g[0]*a0r - g[1]*a0i + g[2]*a1r - g[3]*a1i;
        float n0i = g[0]*a0i + g[1]*a0r + g[2]*a1i + g[3]*a1r;
        float n1r = g[4]*a0r - g[5]*a0i + g[6]*a1r - g[7]*a1i;
        float n1i = g[4]*a0i + g[5]*a0r + g[6]*a1i + g[7]*a1r;
        sre[i0] = n0r; sim[i0] = n0i;
        sre[i1] = n1r; sim[i1] = n1i;
        __syncthreads();
      }
      // CZ-ring sign layer, r = l+1
      int r = l + 1;
      int k0 = t, k1 = t + 128;
      int rot0 = ((k0 << r) | (k0 >> (8 - r))) & 255;
      int rot1 = ((k1 << r) | (k1 >> (8 - r))) & 255;
      if (__popc(k0 & rot0) & 1) { sre[k0] = -sre[k0]; sim[k0] = -sim[k0]; }
      if (__popc(k1 & rot1) & 1) { sre[k1] = -sre[k1]; sim[k1] = -sim[k1]; }
      __syncthreads();
    }
  }

  // probabilities, contract with Z table
  float p0 = sre[t] * sre[t] + sim[t] * sim[t];
  float p1 = sre[t + 128] * sre[t + 128] + sim[t + 128] * sim[t + 128];
  __syncthreads();
  sre[t] = p0; sre[t + 128] = p1;
  __syncthreads();
  if (t < 8) {
    float acc = 0.f;
    for (int k = 0; k < 256; ++k) {
      float z = ((k >> (7 - t)) & 1) ? -1.f : 1.f;
      acc += sre[k] * z;
    }
    xout[b * NH + t] = acc;
  }
}

// ---------------- dense 8->16 -------------------------------------------
__global__ __launch_bounds__(256) void k_wu(const float* __restrict__ xr8,
                                            const float* __restrict__ Wu,
                                            const float* __restrict__ bu,
                                            float* __restrict__ xr16) {
  int b = blockIdx.x * blockDim.x + threadIdx.x;
  if (b >= NB) return;
  float in8[NH];
#pragma unroll
  for (int j = 0; j < NH; ++j) in8[j] = xr8[b * NH + j];
#pragma unroll
  for (int c = 0; c < NC; ++c) {
    float acc = bu[c];
#pragma unroll
    for (int j = 0; j < NH; ++j) acc += in8[j] * Wu[c * NH + j];
    xr16[b * NC + c] = acc;
  }
}

// ---------------- out[2048x4096] = xr16 @ V^T + mean (f32 WMMA, K=16) ----
// grid: x over N tiles of 64 (4 waves x 16), y over M tiles of 16
__global__ __launch_bounds__(128) void k_expand(const float* __restrict__ xr16,
                                                const float* __restrict__ V,
                                                const float* __restrict__ mean,
                                                float* __restrict__ out) {
  int wave = threadIdx.x >> 5;
  int lane = threadIdx.x & 31;
  int half = lane >> 4, l16 = lane & 15;
  int m0 = blockIdx.y * 16;
  int n0 = blockIdx.x * 64 + wave * 16;
  v8f acc = {};
#pragma unroll
  for (int k = 0; k < NC; k += 4) {
    int kk = k + half * 2;
    v2f a, bfr;
    a.x = xr16[(m0 + l16) * NC + kk];
    a.y = xr16[(m0 + l16) * NC + kk + 1];
    bfr.x = V[(size_t)(n0 + l16) * NC + kk];     // B[k,n] = comps[k, n] = V[n, k]
    bfr.y = V[(size_t)(n0 + l16) * NC + kk + 1];
    acc = __builtin_amdgcn_wmma_f32_16x16x4_f32(false, a, false, bfr,
                                                (short)0, acc, false, false);
  }
  float mn = mean[n0 + l16];
#pragma unroll
  for (int v = 0; v < 8; ++v)
    out[(size_t)(m0 + v + 8 * half) * ND + n0 + l16] = acc[v] + mn;
}

// ---------------------------------------------------------------------------
extern "C" void kernel_launch(void* const* d_in, const int* in_sizes, int n_in,
                              void* d_out, int out_size, void* d_ws, size_t ws_size,
                              hipStream_t stream) {
  (void)in_sizes; (void)n_in; (void)out_size; (void)ws_size;
  const float* x     = (const float*)d_in[0];   // 2048*4096
  const float* w1    = (const float*)d_in[1];   // 4*2*2*8*3
  const float* gamma = (const float*)d_in[2];   // 16
  const float* beta  = (const float*)d_in[3];   // 16
  const float* Wd    = (const float*)d_in[4];   // 8x16
  const float* bd    = (const float*)d_in[5];   // 8
  const float* Wu    = (const float*)d_in[6];   // 16x8
  const float* bu    = (const float*)d_in[7];   // 16
  float* out = (float*)d_out;

  float* ws    = (float*)d_ws;
  float* mean  = ws;                    // 4096
  float* V     = mean + ND;             // 4096*16
  float* W     = V + (size_t)ND * NC;   // 4096*16
  float* Y     = W + (size_t)ND * NC;   // 2048*16
  float* xr8   = Y + (size_t)NB * NC;   // 2048*8
  float* xr16  = xr8 + (size_t)NB * NH; // 2048*16
  float* stats = xr16 + (size_t)NB * NC;// 32

  k_colmean<<<ND / 256, 256, 0, stream>>>(x, mean);
  k_initv<<<(ND * NC) / 256, 256, 0, stream>>>(V);

  for (int it = 0; it < NPOW; ++it) {
    k_gemm_xc_v<<<NB / 128, 256, 0, stream>>>(x, mean, V, Y);
    k_gemm_xct_y<<<ND / 128, 256, 0, stream>>>(x, mean, Y, W);
    k_orthonormalize<<<1, 256, 0, stream>>>(W, V);
  }

  k_gemm_xc_v<<<NB / 128, 256, 0, stream>>>(x, mean, V, Y);
  k_bnstats<<<1, 256, 0, stream>>>(Y, stats);
  k_bn_wd<<<NB / 256, 256, 0, stream>>>(Y, stats, gamma, beta, Wd, bd, xr8);

  for (int n = 0; n < 4; ++n)
    k_circuit<<<NB, 128, 0, stream>>>(xr8, w1, xr8, n);

  k_wu<<<NB / 256, 256, 0, stream>>>(xr8, Wu, bu, xr16);
  k_expand<<<dim3(ND / 64, NB / 16), 128, 0, stream>>>(xr16, V, mean, out);
}
